// CrossCondGPT2_83047487635484
// MI455X (gfx1250) — matrix operand order repaired and checked by reading, not compiled, not run on hardware
//
#include <hip/hip_runtime.h>
#include <hip/hip_bf16.h>

typedef __attribute__((ext_vector_type(16))) _Float16 v16h;
typedef __attribute__((ext_vector_type(8)))  _Float16 v8h;
typedef __attribute__((ext_vector_type(8)))  float    v8f;

#define C_DIM  768
#define H_NUM  12
#define D_DIM  64
#define B_NUM  4
#define T_MOT  512
#define ML_DIM 1536   // 3*T
#define L_DIM  1568   // 3*T + 4*8
#define NSEG   8

// ---------- helpers ----------

// Build a v16h fragment from two contiguous 8-half (16B) chunks at p and p+gap.
__device__ __forceinline__ v16h load2x8(const _Float16* p, int gap) {
  v8h lo = *(const v8h*)p;
  v8h hi = *(const v8h*)(p + gap);
  v16h r;
#pragma unroll
  for (int i = 0; i < 8; ++i) { r[i] = lo[i]; r[i + 8] = hi[i]; }
  return r;
}

__device__ __forceinline__ v8f wmma_f16(v16h a, v16h b, v8f c) {
  // D = A(16x32 f16) * B(32x16 f16) + C(16x16 f32)
  return __builtin_amdgcn_wmma_f32_16x16x32_f16(false, a, false, b, (short)0, c,
                                                false, false);
}

__device__ __forceinline__ v8f zero8() {
  v8f z;
#pragma unroll
  for (int i = 0; i < 8; ++i) z[i] = 0.0f;
  return z;
}

// Exact device re-implementation of _build_mask: returns true if (r,c) masked.
__device__ __forceinline__ bool masked_fn(int r, int c, const int* ss, const int* se) {
  if (r >= ML_DIM) return false;                  // text rows see everything
  if (c < ML_DIM) {                               // motion columns
    if (r < T_MOT) {                              // upper-motion rows
      if (c >= T_MOT) return true;
      return c > r;                               // ~tril
    } else if (r < 2 * T_MOT) {                   // lower-motion rows
      int rr = r - T_MOT;
      if (c < T_MOT) return c > rr;               // ~tril
      int cc = (c < 2 * T_MOT) ? (c - T_MOT) : (c - 2 * T_MOT);
      return cc >= rr;                            // ~strictly_lower (both blocks)
    } else {                                      // torso-motion rows
      int rr = r - 2 * T_MOT;
      if (c < T_MOT) return c > rr;               // ~tril
      if (c < 2 * T_MOT) return (c - T_MOT) > rr; // ~tril
      return (c - 2 * T_MOT) >= rr;               // ~strictly_lower
    }
  } else {                                        // text columns
    if (r < T_MOT) return true;                   // upper block: always masked
    int tc = c - ML_DIM;
    int j = tc >> 3;                              // stream (N=8)
    int i = tc & 7;                               // segment index
    int f;
    if (r < 2 * T_MOT) {                          // rows T:2T -> streams {0,2,3}
      if (!(j == 0 || j == 2 || j == 3)) return true;
      f = r - T_MOT;
    } else {                                      // rows 2T:3T -> streams {1,2,3}
      if (!(j == 1 || j == 2 || j == 3)) return true;
      f = r - 2 * T_MOT;
    }
    return !((f >= ss[i]) && (f < se[i]));
  }
}

// ---------- kernels ----------

__global__ __launch_bounds__(256) void cvt_f32_f16(const float* __restrict__ in,
                                                   _Float16* __restrict__ out, int n) {
  int i = blockIdx.x * 256 + threadIdx.x;
  if (i < n) out[i] = (_Float16)in[i];
}

// y[m, o] = sum_c A[m,c] * W[o,c] + bias[o]   (A: [M,768] f16, W: [768,768] f16)
// One wave computes a 16(M) x 64(O) tile; 24 K-steps of 32.
// mode 0: f16 out at [m*768+o]; mode 1: f16 out transposed [b,h,d,l]; mode 2: f32 out.
__global__ __launch_bounds__(32) void gemm_wmma(const _Float16* __restrict__ A,
                                                const _Float16* __restrict__ W,
                                                const float* __restrict__ bias,
                                                _Float16* __restrict__ out16,
                                                float* __restrict__ out32,
                                                int mode) {
  const int lane = threadIdx.x;
  const int lr = lane & 15;
  const int hg = lane >> 4;
  const int m0 = blockIdx.y * 16;
  const int o0 = blockIdx.x * 64;

  v8f acc[4];
#pragma unroll
  for (int t = 0; t < 4; ++t) acc[t] = zero8();

  const _Float16* arow = A + (size_t)(m0 + lr) * C_DIM;
  for (int k0 = 0; k0 < C_DIM; k0 += 32) {
    v16h a = load2x8(arow + k0 + 8 * hg, 16);   // A-frag layout (16-bit 16x32)
#pragma unroll
    for (int t = 0; t < 4; ++t) {
      const _Float16* wrow = W + (size_t)(o0 + 16 * t + lr) * C_DIM + k0 + 8 * hg;
      v16h b = load2x8(wrow, 16);               // B-frag: same K order as A-frag
      acc[t] = wmma_f16(a, b, acc[t]);
    }
  }

#pragma unroll
  for (int t = 0; t < 4; ++t) {
    const int o = o0 + 16 * t + lr;
    const float bo = bias[o];
#pragma unroll
    for (int j = 0; j < 8; ++j) {
      const int m = m0 + j + 8 * hg;            // C-layout: lane holds (M=j+8*hg, N=lr)
      const float v = acc[t][j] + bo;
      if (mode == 0) {
        out16[(size_t)m * C_DIM + o] = (_Float16)v;
      } else if (mode == 1) {                   // store V transposed: [B,H,D,L]
        const int bb = m / L_DIM;
        const int l  = m - bb * L_DIM;
        const int h  = o >> 6;
        const int d  = o & 63;
        out16[(((size_t)(bb * H_NUM + h)) * D_DIM + d) * L_DIM + l] = (_Float16)v;
      } else {
        out32[(size_t)m * C_DIM + o] = v;
      }
    }
  }
}

// Flash attention: grid (L/16, H, B), one wave per block.
__global__ __launch_bounds__(32) void attn_wmma(const _Float16* __restrict__ Qh,
                                                const _Float16* __restrict__ Kh,
                                                const _Float16* __restrict__ Vt,
                                                const int* __restrict__ seg_s,
                                                const int* __restrict__ seg_e,
                                                _Float16* __restrict__ attout) {
  const int lane = threadIdx.x;
  const int lr = lane & 15;
  const int hg = lane >> 4;
  const int m0 = blockIdx.x * 16;
  const int h  = blockIdx.y;
  const int b  = blockIdx.z;
  const float scale = 0.125f; // 1/sqrt(64)

  __shared__ __align__(16) _Float16 lds[16 * 32];

  int ss[NSEG], se[NSEG];
#pragma unroll
  for (int i = 0; i < NSEG; ++i) { ss[i] = seg_s[b * NSEG + i]; se[i] = seg_e[b * NSEG + i]; }

  // Q fragments: rows m0..m0+15, K-dim d=0..63 in two 32-wide A-fragments.
  const _Float16* qrow = Qh + ((size_t)(b * L_DIM + m0 + lr)) * C_DIM + h * D_DIM;
  const v16h aq0 = load2x8(qrow + 8 * hg, 16);        // d 0..31
  const v16h aq1 = load2x8(qrow + 32 + 8 * hg, 16);   // d 32..63

  float rm[8], rs[8];
#pragma unroll
  for (int j = 0; j < 8; ++j) { rm[j] = -1e30f; rs[j] = 0.0f; }
  v8f oacc[4];
#pragma unroll
  for (int t = 0; t < 4; ++t) oacc[t] = zero8();

  const _Float16* vbase = Vt + ((size_t)(b * H_NUM + h)) * D_DIM * L_DIM;

  for (int n0 = 0; n0 < L_DIM; n0 += 32) {
    // ---- S = Q K^T for 16x32 key block (two 16x16 tiles) ----
    const _Float16* krow_lo = Kh + ((size_t)(b * L_DIM + n0 + lr)) * C_DIM + h * D_DIM + 8 * hg;
    const _Float16* krow_hi = krow_lo + (size_t)16 * C_DIM;
    v16h bk0 = load2x8(krow_lo, 16);       // keys n0..n0+15, d 0..31
    v16h bk1 = load2x8(krow_lo + 32, 16);  // keys n0..n0+15, d 32..63
    v16h bk2 = load2x8(krow_hi, 16);       // keys n0+16..n0+31, d 0..31
    v16h bk3 = load2x8(krow_hi + 32, 16);

    v8f s_lo = wmma_f16(aq1, bk1, wmma_f16(aq0, bk0, zero8()));
    v8f s_hi = wmma_f16(aq1, bk3, wmma_f16(aq0, bk2, zero8()));

    // ---- mask + online softmax (per-row stats across 16 lanes) ----
    float pl[8], ph[8], al[8];
#pragma unroll
    for (int j = 0; j < 8; ++j) {
      const int r  = m0 + j + 8 * hg;
      const int cl = n0 + lr;
      const int ch = n0 + 16 + lr;
      const bool mlq = masked_fn(r, cl, ss, se);
      const bool mhq = masked_fn(r, ch, ss, se);
      float sl = mlq ? -1e30f : s_lo[j] * scale;
      float sh = mhq ? -1e30f : s_hi[j] * scale;
      float tmax = fmaxf(sl, sh);
#pragma unroll
      for (int off = 1; off < 16; off <<= 1) tmax = fmaxf(tmax, __shfl_xor(tmax, off, 32));
      const float nm = fmaxf(rm[j], tmax);
      const float a  = __expf(rm[j] - nm);       // finite sentinel keeps this well-defined
      float plj = mlq ? 0.0f : __expf(sl - nm);
      float phj = mhq ? 0.0f : __expf(sh - nm);
      float tsum = plj + phj;
#pragma unroll
      for (int off = 1; off < 16; off <<= 1) tsum += __shfl_xor(tsum, off, 32);
      rs[j] = rs[j] * a + tsum;
      rm[j] = nm;
      al[j] = a; pl[j] = plj; ph[j] = phj;
    }

    // rescale running output
#pragma unroll
    for (int t = 0; t < 4; ++t)
#pragma unroll
      for (int j = 0; j < 8; ++j) oacc[t][j] *= al[j];

    // ---- transpose P (C-layout) -> A-layout via LDS ----
    __syncthreads();
#pragma unroll
    for (int j = 0; j < 8; ++j) {
      lds[(j + 8 * hg) * 32 + lr]      = (_Float16)pl[j];
      lds[(j + 8 * hg) * 32 + 16 + lr] = (_Float16)ph[j];
    }
    __syncthreads();
    v16h ap = load2x8(&lds[lr * 32 + 8 * hg], 16);

    // ---- O += P * V (4 d-subtiles) ----
#pragma unroll
    for (int t = 0; t < 4; ++t) {
      const _Float16* vrow = vbase + (size_t)(16 * t + lr) * L_DIM + n0 + 8 * hg;
      v16h bv = load2x8(vrow, 16);
      oacc[t] = wmma_f16(ap, bv, oacc[t]);
    }
  }

  // ---- normalize and store att output as f16 [B,L,C] ----
#pragma unroll
  for (int j = 0; j < 8; ++j) {
    const float inv = 1.0f / rs[j];
    const size_t row = (size_t)(b * L_DIM + m0 + j + 8 * hg) * C_DIM + h * D_DIM;
#pragma unroll
    for (int t = 0; t < 4; ++t)
      attout[row + 16 * t + lr] = (_Float16)(oacc[t][j] * inv);
  }
}

// ---------- launch ----------

extern "C" void kernel_launch(void* const* d_in, const int* in_sizes, int n_in,
                              void* d_out, int out_size, void* d_ws, size_t ws_size,
                              hipStream_t stream) {
  const float* x  = (const float*)d_in[0];
  const float* Wq = (const float*)d_in[1];
  const float* bq = (const float*)d_in[2];
  const float* Wk = (const float*)d_in[3];
  const float* bk = (const float*)d_in[4];
  const float* Wv = (const float*)d_in[5];
  const float* bv = (const float*)d_in[6];
  const float* Wp = (const float*)d_in[7];
  const float* bp = (const float*)d_in[8];
  const int* seg_s = (const int*)d_in[9];
  const int* seg_e = (const int*)d_in[10];

  const size_t NX = (size_t)B_NUM * L_DIM * C_DIM;   // 4,816,896
  const size_t NW = (size_t)C_DIM * C_DIM;           // 589,824

  _Float16* p   = (_Float16*)d_ws;
  _Float16* x_h  = p; p += NX;
  _Float16* wq_h = p; p += NW;
  _Float16* wk_h = p; p += NW;
  _Float16* wv_h = p; p += NW;
  _Float16* wp_h = p; p += NW;
  _Float16* q_h  = p; p += NX;
  _Float16* k_h  = p; p += NX;
  _Float16* vt_h = p; p += NX;   // [B,H,D,L]
  _Float16* at_h = p; p += NX;

  cvt_f32_f16<<<(int)((NX + 255) / 256), 256, 0, stream>>>(x,  x_h,  (int)NX);
  cvt_f32_f16<<<(int)((NW + 255) / 256), 256, 0, stream>>>(Wq, wq_h, (int)NW);
  cvt_f32_f16<<<(int)((NW + 255) / 256), 256, 0, stream>>>(Wk, wk_h, (int)NW);
  cvt_f32_f16<<<(int)((NW + 255) / 256), 256, 0, stream>>>(Wv, wv_h, (int)NW);
  cvt_f32_f16<<<(int)((NW + 255) / 256), 256, 0, stream>>>(Wp, wp_h, (int)NW);

  dim3 ggrid(C_DIM / 64, (B_NUM * L_DIM) / 16);      // (12, 392)
  gemm_wmma<<<ggrid, 32, 0, stream>>>(x_h, wq_h, bq, q_h, nullptr, 0);
  gemm_wmma<<<ggrid, 32, 0, stream>>>(x_h, wk_h, bk, k_h, nullptr, 0);
  gemm_wmma<<<ggrid, 32, 0, stream>>>(x_h, wv_h, bv, vt_h, nullptr, 1);

  dim3 agrid(L_DIM / 16, H_NUM, B_NUM);              // (98, 12, 4)
  attn_wmma<<<agrid, 32, 0, stream>>>(q_h, k_h, vt_h, seg_s, seg_e, at_h);

  gemm_wmma<<<ggrid, 32, 0, stream>>>(at_h, wp_h, bp, nullptr, (float*)d_out, 2);
}
